// DiagonalEmbedder_66434554134953
// MI455X (gfx1250) — matrix operand order repaired and checked by reading, not compiled, not run on hardware
//
#include <hip/hip_runtime.h>

typedef __attribute__((ext_vector_type(16))) _Float16 v16h;
typedef __attribute__((ext_vector_type(8)))  float    v8f;
typedef __attribute__((ext_vector_type(4)))  int      v4i;

#define BATCH 4096
#define NIN   8192
#define DIM   128
#define KDIM  256          // [x^2 | x] and [inv | -2*mean*inv]
#define LDA   264          // KDIM + 8 halfs pad: 528B row stride -> conflict-free b128 frag loads
#define MT    256          // workgroup M tile
#define NT    128          // workgroup N tile
#define PD_THR 1e-6f

#if __has_builtin(__builtin_amdgcn_global_load_async_to_lds_b128) && \
    __has_builtin(__builtin_amdgcn_s_wait_asynccnt)
#define USE_ASYNC_LDS 1
typedef __attribute__((address_space(1))) v4i* as1_v4i;
typedef __attribute__((address_space(3))) v4i* as3_v4i;
#else
#define USE_ASYNC_LDS 0
#endif

// ---------------------------------------------------------------------------
// Prep 1: A[b,:] = [x^2 (128) | x (128)] in f16, row-major 4096 x 256
// ---------------------------------------------------------------------------
__global__ __launch_bounds__(256) void build_a_kernel(
    const float* __restrict__ x, _Float16* __restrict__ A) {
  int idx = blockIdx.x * 256 + threadIdx.x;      // 4096*128 threads
  if (idx >= BATCH * DIM) return;
  int b = idx >> 7;
  int d = idx & (DIM - 1);
  float xv = x[idx];
  A[(size_t)b * KDIM + d]        = (_Float16)(xv * xv);
  A[(size_t)b * KDIM + DIM + d]  = (_Float16)xv;
}

// ---------------------------------------------------------------------------
// Prep 2: W[i,:] = [inv (128) | -2*mean*inv (128)] f16, row-major 8192 x 256
//         bias[i] = sum(mean^2*inv) + sum(log d) + sum(inv) - DIM
// ---------------------------------------------------------------------------
__global__ __launch_bounds__(128) void build_w_kernel(
    const float* __restrict__ mean, const float* __restrict__ diag,
    _Float16* __restrict__ W, float* __restrict__ bias) {
  const int i = blockIdx.x;
  const int d = threadIdx.x;
  float dg  = diag[(size_t)i * DIM + d];
  float dc  = fmaxf(dg, PD_THR);
  float inv = 1.0f / dc;
  float m   = mean[(size_t)i * DIM + d];
  W[(size_t)i * KDIM + d]       = (_Float16)inv;
  W[(size_t)i * KDIM + DIM + d] = (_Float16)(-2.0f * m * inv);

  __shared__ float red[128];
  red[d] = m * m * inv + __logf(dc) + inv;
  __syncthreads();
  #pragma unroll
  for (int s = 64; s > 0; s >>= 1) {
    if (d < s) red[d] += red[d + s];
    __syncthreads();
  }
  if (d == 0) bias[i] = red[0] - (float)DIM;
}

// ---------------------------------------------------------------------------
// Fragment loaders (LDS -> VGPR, layouts per CDNA5 ISA 7.12.2)
// ---------------------------------------------------------------------------
__device__ __forceinline__ v16h load_afrag(const _Float16* p) {
  // 16x32 f16 A: per lane two 16B chunks, 32B apart (K and K+16)
  union { uint4 u[2]; v16h h; } t;
  t.u[0] = *(const uint4*)p;
  t.u[1] = *(const uint4*)(p + 16);
  return t.h;
}
__device__ __forceinline__ v16h load_bfrag(const _Float16* p) {
  // 32x16 f16 B, column-major: per lane two 16B chunks, 32B apart (K and K+16)
  union { uint4 u[2]; v16h h; } t;
  t.u[0] = *(const uint4*)p;
  t.u[1] = *(const uint4*)(p + 16);
  return t.h;
}

// ---------------------------------------------------------------------------
// GEMM: out[b,i] = 0.5*( A(4096x256) . W(8192x256)^T + bias[i] )
// 256x128 tile / 256-thread block; 8 waves in 4(M) x 2(N); each wave 64x64
// (4x4 sub-tiles of 16x16): 16 WMMAs per 8 fragment loads per k-step.
// Full K staged in LDS (~198 KB) via async-to-LDS; NT stores for the 128MB
// output stream.
// ---------------------------------------------------------------------------
__global__ __launch_bounds__(256) void kl_gemm_kernel(
    const _Float16* __restrict__ A, const _Float16* __restrict__ W,
    const float* __restrict__ bias, float* __restrict__ out) {
  __shared__ _Float16 As[MT * LDA];    // 256 rows, ~132 KB
  __shared__ _Float16 Bs[NT * LDA];    // 128 rows, ~66 KB

  const int tid = threadIdx.x;
  const int bn  = blockIdx.x;          // N tile (n_in)
  const int bm  = blockIdx.y;          // M tile (batch)

  // Tiles are contiguous blocks in global (full-K rows, row-major).
  const uint4* gA = (const uint4*)(A + (size_t)bm * MT * KDIM);   // 8192 chunks
  const uint4* gB = (const uint4*)(W + (size_t)bn * NT * KDIM);   // 4096 chunks
  #pragma unroll
  for (int c = 0; c < 32; ++c) {
    int e  = tid + c * 256;
    int r  = e >> 5, kc = e & 31;      // 32 x 16B chunks per 256-half row
    _Float16* dstA = As + (size_t)r * LDA + kc * 8;
#if USE_ASYNC_LDS
    __builtin_amdgcn_global_load_async_to_lds_b128(
        (as1_v4i)(gA + e), (as3_v4i)dstA, 0, 0);
#else
    *(uint4*)dstA = gA[e];
#endif
  }
  #pragma unroll
  for (int c = 0; c < 16; ++c) {
    int e  = tid + c * 256;
    int r  = e >> 5, kc = e & 31;
    _Float16* dstB = Bs + (size_t)r * LDA + kc * 8;
#if USE_ASYNC_LDS
    __builtin_amdgcn_global_load_async_to_lds_b128(
        (as1_v4i)(gB + e), (as3_v4i)dstB, 0, 0);
#else
    *(uint4*)dstB = gB[e];
#endif
  }
#if USE_ASYNC_LDS
  __builtin_amdgcn_s_wait_asynccnt(0);
#endif
  __syncthreads();

  const int lane = tid & 31;
  const int w    = tid >> 5;
  const int wm   = (w >> 1) * 64;      // wave M offset in tile (0..192)
  const int wn   = (w & 1) * 64;       // wave N offset in tile (0 or 64)
  const int half = lane >> 4;          // lane group 0/1
  const int l16  = lane & 15;

  // Per-lane fragment base addresses (k and sub-tile offsets added per step)
  const _Float16* aBase = As + (size_t)(wm + l16) * LDA + half * 8;
  const _Float16* bBase = Bs + (size_t)(wn + l16) * LDA + half * 8;

  v8f acc[4][4];
  #pragma unroll
  for (int i = 0; i < 4; ++i)
    #pragma unroll
    for (int j = 0; j < 4; ++j) acc[i][j] = (v8f){};

  #pragma unroll
  for (int ks = 0; ks < 8; ++ks) {
    const int kk = ks * 32;
    v16h bf[4], af[4];
    #pragma unroll
    for (int j = 0; j < 4; ++j) bf[j] = load_bfrag(bBase + j * 16 * LDA + kk);
    #pragma unroll
    for (int i = 0; i < 4; ++i) af[i] = load_afrag(aBase + i * 16 * LDA + kk);
    #pragma unroll
    for (int i = 0; i < 4; ++i) {
      #pragma unroll
      for (int j = 0; j < 4; ++j) {
        acc[i][j] = __builtin_amdgcn_wmma_f32_16x16x32_f16(
            false, af[i], false, bf[j], (short)0, acc[i][j], false, false);
      }
    }
  }

  // Epilogue: out = 0.5*(acc + bias[col]); C/D layout: VGPR v -> M = v + half*8
  #pragma unroll
  for (int j = 0; j < 4; ++j) {
    const int col = bn * NT + wn + j * 16 + l16;
    const float bv = bias[col];
    #pragma unroll
    for (int i = 0; i < 4; ++i) {
      const int rbase = bm * MT + wm + i * 16 + half * 8;
      #pragma unroll
      for (int v = 0; v < 8; ++v) {
        __builtin_nontemporal_store(0.5f * (acc[i][j][v] + bv),
                                    out + (size_t)(rbase + v) * NIN + col);
      }
    }
  }
}

// ---------------------------------------------------------------------------
extern "C" void kernel_launch(void* const* d_in, const int* in_sizes, int n_in,
                              void* d_out, int out_size, void* d_ws, size_t ws_size,
                              hipStream_t stream) {
  const float* x    = (const float*)d_in[0];   // (4096, 128)
  const float* mean = (const float*)d_in[1];   // (8192, 128)
  const float* diag = (const float*)d_in[2];   // (8192, 128)
  float* out = (float*)d_out;                  // (4096, 8192)

  char* ws = (char*)d_ws;
  _Float16* Abuf = (_Float16*)ws;                                   // 2 MB
  _Float16* Wbuf = (_Float16*)(ws + (size_t)BATCH * KDIM * 2);      // 4 MB
  float*    bias = (float*)(ws + (size_t)BATCH * KDIM * 2
                               + (size_t)NIN * KDIM * 2);           // 32 KB

  build_a_kernel<<<(BATCH * DIM) / 256, 256, 0, stream>>>(x, Abuf);
  build_w_kernel<<<NIN, DIM, 0, stream>>>(mean, diag, Wbuf, bias);

  dim3 grid(NIN / NT, BATCH / MT);     // (64, 16)
  kl_gemm_kernel<<<grid, 256, 0, stream>>>(Abuf, Wbuf, bias, out);
}
